// MoEWrapper_18004502905406
// MI455X (gfx1250) — matrix operand order repaired and checked by reading, not compiled, not run on hardware
//
#include <hip/hip_runtime.h>

#define B_TOK 8192
#define DIN   1024
#define DOUT  1024
#define NE    8
#define NH    128
#define BK    2048   // tokens selected per expert = ceil(K*B/E)

typedef __bf16 bf16;
typedef bf16  v16bf __attribute__((ext_vector_type(16)));
typedef float v8f   __attribute__((ext_vector_type(8)));

union FragAB { v16bf v; uint4 q[2]; };

// ---------------------------------------------------------------- utilities
__global__ void zero_f32(float* __restrict__ p, long n) {
  long i = (long)blockIdx.x * blockDim.x + threadIdx.x;
  long st = (long)gridDim.x * blockDim.x;
  for (; i < n; i += st) p[i] = 0.f;
}

__global__ void cvt_to_bf16(const float* __restrict__ s, bf16* __restrict__ d, long n) {
  long i = (long)blockIdx.x * blockDim.x + threadIdx.x;
  long st = (long)gridDim.x * blockDim.x;
  for (; i < n; i += st) d[i] = (bf16)s[i];
}

// f32 [K][N] -> bf16 [N][K] (per z-slice), 32x32 LDS-tiled, coalesced both sides
__global__ __launch_bounds__(256) void transpose_cvt(
    const float* __restrict__ src, bf16* __restrict__ dst, int K, int N) {
  __shared__ float s[32][33];
  const int tid = threadIdx.x;
  const int kb = blockIdx.x * 32, nb = blockIdx.y * 32;
  const float* sp = src + (size_t)blockIdx.z * K * N;
  bf16*        dp = dst + (size_t)blockIdx.z * K * N;
  #pragma unroll
  for (int i = 0; i < 4; ++i) {
    int lin = i * 256 + tid;
    int kr = lin >> 5, nc = lin & 31;
    s[kr][nc] = sp[(size_t)(kb + kr) * N + nb + nc];
  }
  __syncthreads();
  #pragma unroll
  for (int i = 0; i < 4; ++i) {
    int lin = i * 256 + tid;
    int nr = lin >> 5, kc = lin & 31;
    dp[(size_t)(nb + nr) * K + kb + kc] = (bf16)s[kc][nr];
  }
}

// ------------------------------------------------------------- router GEMM
// h = relu((x@w1 + b1) * relu(x@wg + bg)) ; one block = 128 token rows, N=128
// weights pre-transposed: w1t/wgt are [NH][DIN] bf16
__global__ __launch_bounds__(256) void router_gemm(
    const bf16* __restrict__ xb, const bf16* __restrict__ w1t,
    const bf16* __restrict__ wgt, const float* __restrict__ b1,
    const float* __restrict__ bg, float* __restrict__ hbuf) {
  __shared__ bf16 sA [128][32];   // token rows, K-major
  __shared__ bf16 sW1[128][32];   // [n][k]
  __shared__ bf16 sWg[128][32];
  const int tid = threadIdx.x, lane = tid & 31, wave = tid >> 5;
  const int mbase = blockIdx.x * 128;
  const int row2 = tid >> 1, half2 = (tid & 1) * 16;

  v8f acc1[8], acc2[8];
  for (int i = 0; i < 8; ++i)
    for (int j = 0; j < 8; ++j) { acc1[i][j] = 0.f; acc2[i][j] = 0.f; }

  for (int kt = 0; kt < DIN; kt += 32) {
    // stage A: 128x32 bf16, 2 threads per row, b128 stores only
    {
      const uint4* g = (const uint4*)(xb + (size_t)(mbase + row2) * DIN + kt + half2);
      uint4 a0 = g[0], a1 = g[1];
      *(uint4*)&sA[row2][half2]     = a0;
      *(uint4*)&sA[row2][half2 + 8] = a1;
    }
    // stage W1/Wg tiles (already [n][k]); plain b128 copies
    {
      const uint4* g1 = (const uint4*)(w1t + (size_t)row2 * DIN + kt + half2);
      uint4 a0 = g1[0], a1 = g1[1];
      *(uint4*)&sW1[row2][half2]     = a0;
      *(uint4*)&sW1[row2][half2 + 8] = a1;
      const uint4* gg = (const uint4*)(wgt + (size_t)row2 * DIN + kt + half2);
      uint4 b0 = gg[0], b1v = gg[1];
      *(uint4*)&sWg[row2][half2]     = b0;
      *(uint4*)&sWg[row2][half2 + 8] = b1v;
    }
    __syncthreads();

    // A fragment for this wave's 16 rows (ISA 16-bit A 16x32 layout)
    FragAB fa;
    {
      int row = wave * 16 + (lane & 15);
      int k0  = (lane < 16) ? 0 : 8;
      const bf16* p = &sA[row][0];
      fa.q[0] = *(const uint4*)(p + k0);
      fa.q[1] = *(const uint4*)(p + k0 + 16);
    }
    #pragma unroll
    for (int nf = 0; nf < 8; ++nf) {
      int col = nf * 16 + (lane & 15);
      int k0  = (lane < 16) ? 0 : 8;
      FragAB fb;
      const bf16* p1 = &sW1[col][0] + k0;
      fb.q[0] = *(const uint4*)p1;
      fb.q[1] = *(const uint4*)(p1 + 16);
      acc1[nf] = __builtin_amdgcn_wmma_f32_16x16x32_bf16(
          false, fa.v, false, fb.v, (short)0, acc1[nf], false, false);
      const bf16* pg = &sWg[col][0] + k0;
      fb.q[0] = *(const uint4*)pg;
      fb.q[1] = *(const uint4*)(pg + 16);
      acc2[nf] = __builtin_amdgcn_wmma_f32_16x16x32_bf16(
          false, fa.v, false, fb.v, (short)0, acc2[nf], false, false);
    }
    __syncthreads();
  }

  // epilogue: GLU + ReLU, write h
  int rbase = mbase + wave * 16 + ((lane < 16) ? 0 : 8);
  int nlo = lane & 15;
  for (int nf = 0; nf < 8; ++nf) {
    int n = nf * 16 + nlo;
    float bb1 = b1[n], bbg = bg[n];
    #pragma unroll
    for (int r = 0; r < 8; ++r) {
      float v1 = acc1[nf][r] + bb1;
      float v2 = acc2[nf][r] + bbg;
      float hv = fmaxf(v1 * fmaxf(v2, 0.f), 0.f);
      hbuf[(size_t)(rbase + r) * NH + n] = hv;
    }
  }
}

// ------------------------------------------------------------ tiny logits
__global__ __launch_bounds__(256) void logits_kernel(
    const float* __restrict__ hbuf, const float* __restrict__ w2,
    const float* __restrict__ b2, float* __restrict__ l) {
  __shared__ float sw2[NH * NE];
  __shared__ float sb2[NE];
  for (int i = threadIdx.x; i < NH * NE; i += blockDim.x) sw2[i] = w2[i];
  if (threadIdx.x < NE) sb2[threadIdx.x] = b2[threadIdx.x];
  __syncthreads();
  int b = blockIdx.x * blockDim.x + threadIdx.x;
  if (b >= B_TOK) return;
  float acc[NE];
  for (int e = 0; e < NE; ++e) acc[e] = sb2[e];
  const float* hr = hbuf + (size_t)b * NH;
  for (int j = 0; j < NH; ++j) {
    float hv = hr[j];
    #pragma unroll
    for (int e = 0; e < NE; ++e) acc[e] += hv * sw2[j * NE + e];
  }
  for (int e = 0; e < NE; ++e) l[(size_t)b * NE + e] = acc[e];
}

// ----------------------------------------- softmax over BATCH dim, per expert
__global__ __launch_bounds__(256) void softmax_batch(float* __restrict__ l) {
  __shared__ float red[256];
  const int e = blockIdx.x, tid = threadIdx.x;
  float vals[32];
  float m = -3.4e38f;
  for (int i = 0; i < 32; ++i) {
    vals[i] = l[(size_t)(tid + i * 256) * NE + e];
    m = fmaxf(m, vals[i]);
  }
  red[tid] = m; __syncthreads();
  for (int s = 128; s > 0; s >>= 1) { if (tid < s) red[tid] = fmaxf(red[tid], red[tid + s]); __syncthreads(); }
  m = red[0]; __syncthreads();
  float sum = 0.f;
  for (int i = 0; i < 32; ++i) { vals[i] = __expf(vals[i] - m); sum += vals[i]; }
  red[tid] = sum; __syncthreads();
  for (int s = 128; s > 0; s >>= 1) { if (tid < s) red[tid] += red[tid + s]; __syncthreads(); }
  float inv = 1.f / red[0];
  for (int i = 0; i < 32; ++i) l[(size_t)(tid + i * 256) * NE + e] = vals[i] * inv;
}

// -------------------- per-expert top-2048 (bitonic, val desc / idx asc) + nws
__global__ __launch_bounds__(1024) void topk_kernel(
    const float* __restrict__ l, int* __restrict__ ib, float* __restrict__ nws) {
  __shared__ float sv[8192];
  __shared__ int   si[8192];
  const int e = blockIdx.x, tid = threadIdx.x;
  for (int i = tid; i < 8192; i += 1024) { sv[i] = l[(size_t)i * NE + e]; si[i] = i; }
  __syncthreads();
  for (int k = 2; k <= 8192; k <<= 1) {
    for (int j = k >> 1; j > 0; j >>= 1) {
      for (int p = tid; p < 8192; p += 1024) {
        int q = p ^ j;
        if (q > p) {
          bool up = ((p & k) == 0);
          float av = sv[p], bv = sv[q];
          int   ai = si[p], bi = si[q];
          bool b_first = (bv > av) || (bv == av && bi < ai);  // b should rank earlier
          if (up == b_first) { sv[p] = bv; si[p] = bi; sv[q] = av; si[q] = ai; }
        }
      }
      __syncthreads();
    }
  }
  // renorm softmax over the 2048 selected (sorted descending -> max = sv[0])
  float m = sv[0];
  if (tid == 0) sv[4096] = 0.f;   // scratch in dead region
  __syncthreads();
  float e0 = __expf(sv[tid] - m);
  float e1 = __expf(sv[tid + 1024] - m);
  atomicAdd(&sv[4096], e0 + e1);
  __syncthreads();
  float inv = 1.f / sv[4096];
  nws[(size_t)e * BK + tid]        = e0 * inv;
  nws[(size_t)e * BK + tid + 1024] = e1 * inv;
  ib [(size_t)e * BK + tid]        = si[tid];
  ib [(size_t)e * BK + tid + 1024] = si[tid + 1024];
}

// ------------------------------------------------- gather + expert GEMM + scatter
// expert weights pre-transposed: webt is [E][DOUT][DIN] bf16
__global__ __launch_bounds__(256) void expert_gemm(
    const bf16* __restrict__ xb, const bf16* __restrict__ webt,
    const float* __restrict__ be, const int* __restrict__ ib,
    const float* __restrict__ nws, float* __restrict__ out,
    float* __restrict__ ts) {
  __shared__ bf16 sA[128][32];
  __shared__ bf16 sB[128][32];    // [n][k]
  __shared__ int   rows[128];
  __shared__ float scl[128];
  const int tid = threadIdx.x, lane = tid & 31, wave = tid >> 5;
  const int mt = blockIdx.x, nt = blockIdx.y, e = blockIdx.z;
  const int row2 = tid >> 1, half2 = (tid & 1) * 16;

  if (tid < 128) {
    int kk = mt * 128 + tid;
    rows[tid] = ib [(size_t)e * BK + kk];
    scl[tid]  = nws[(size_t)e * BK + kk];
  }
  __syncthreads();

  v8f acc[8];
  for (int i = 0; i < 8; ++i)
    for (int j = 0; j < 8; ++j) acc[i][j] = 0.f;

  const bf16* wbase = webt + (size_t)e * DIN * DOUT + (size_t)(nt * 128) * DIN;

  for (int kt = 0; kt < DIN; kt += 32) {
    // gather A rows (selected tokens), b128 stores
    {
      const uint4* g = (const uint4*)(xb + (size_t)rows[row2] * DIN + kt + half2);
      uint4 a0 = g[0], a1 = g[1];
      *(uint4*)&sA[row2][half2]     = a0;
      *(uint4*)&sA[row2][half2 + 8] = a1;
    }
    // stage we^T tile: plain b128 copies (no transpose needed)
    {
      const uint4* g = (const uint4*)(wbase + (size_t)row2 * DIN + kt + half2);
      uint4 b0 = g[0], b1v = g[1];
      *(uint4*)&sB[row2][half2]     = b0;
      *(uint4*)&sB[row2][half2 + 8] = b1v;
    }
    __syncthreads();

    FragAB fa;
    {
      int row = wave * 16 + (lane & 15);
      int k0  = (lane < 16) ? 0 : 8;
      const bf16* p = &sA[row][0];
      fa.q[0] = *(const uint4*)(p + k0);
      fa.q[1] = *(const uint4*)(p + k0 + 16);
    }
    #pragma unroll
    for (int nf = 0; nf < 8; ++nf) {
      int col = nf * 16 + (lane & 15);
      int k0  = (lane < 16) ? 0 : 8;
      FragAB fb;
      const bf16* p = &sB[col][0] + k0;
      fb.q[0] = *(const uint4*)p;
      fb.q[1] = *(const uint4*)(p + 16);
      acc[nf] = __builtin_amdgcn_wmma_f32_16x16x32_bf16(
          false, fa.v, false, fb.v, (short)0, acc[nf], false, false);
    }
    __syncthreads();
  }

  // epilogue: (acc + be) * nws -> scatter-set into out, atomic-add temp_sum
  int mloc = wave * 16 + ((lane < 16) ? 0 : 8);
  int nlo  = lane & 15;
  for (int nf = 0; nf < 8; ++nf) {
    int n = nt * 128 + nf * 16 + nlo;
    float bias = be[(size_t)e * DOUT + n];
    #pragma unroll
    for (int r = 0; r < 8; ++r) {
      int ml  = mloc + r;
      int tok = rows[ml];
      float v = (acc[nf][r] + bias) * scl[ml];
      out[(size_t)tok * (NE * DOUT) + (size_t)e * DOUT + n] = v;
      atomicAdd(&ts[(size_t)tok * DOUT + n], v);
    }
  }
}

// ------------------------------------------------- broadcast temp_sum combine
__global__ void combine_kernel(float* __restrict__ out, const float* __restrict__ ts) {
  size_t i = (size_t)blockIdx.x * blockDim.x + threadIdx.x;
  size_t st = (size_t)gridDim.x * blockDim.x;
  const size_t total = (size_t)B_TOK * NE * DOUT;
  for (; i < total; i += st) {
    size_t b = i >> 13;        // / (NE*DOUT)
    size_t n = i & (DOUT - 1); // col within expert block
    out[i] += ts[(b << 10) + n];
  }
}

// --------------------------------------------------------------------------
extern "C" void kernel_launch(void* const* d_in, const int* in_sizes, int n_in,
                              void* d_out, int out_size, void* d_ws, size_t ws_size,
                              hipStream_t stream) {
  (void)in_sizes; (void)n_in; (void)out_size; (void)ws_size;
  const float* x  = (const float*)d_in[0];
  const float* w1 = (const float*)d_in[1];
  const float* b1 = (const float*)d_in[2];
  const float* wg = (const float*)d_in[3];
  const float* bg = (const float*)d_in[4];
  const float* w2 = (const float*)d_in[5];
  const float* b2 = (const float*)d_in[6];
  const float* we = (const float*)d_in[7];
  const float* be = (const float*)d_in[8];
  float* out = (float*)d_out;
  char*  ws  = (char*)d_ws;

  // workspace layout (all offsets 256B aligned), total ~69 MB
  bf16*  xb   = (bf16*) (ws + 0);                       // 16 MB
  bf16*  webt = (bf16*) (ws + 16777216);                // 16 MB  [E][DOUT][DIN]
  bf16*  w1t  = (bf16*) (ws + 33554432);                // 256 KB [NH][DIN]
  bf16*  wgt  = (bf16*) (ws + 33816576);                // 256 KB [NH][DIN]
  float* hbuf = (float*)(ws + 34078720);                // 4 MB
  float* lbuf = (float*)(ws + 38273024);                // 256 KB
  int*   ibp  = (int*)  (ws + 38535168);                // 64 KB
  float* nwsp = (float*)(ws + 38600704);                // 64 KB
  float* ts   = (float*)(ws + 38666240);                // 32 MB

  zero_f32<<<4096, 256, 0, stream>>>(out, (long)B_TOK * NE * DOUT);
  zero_f32<<<2048, 256, 0, stream>>>(ts,  (long)B_TOK * DOUT);

  cvt_to_bf16<<<2048, 256, 0, stream>>>(x, xb, (long)B_TOK * DIN);
  transpose_cvt<<<dim3(DIN / 32, DOUT / 32, NE), 256, 0, stream>>>(we, webt, DIN, DOUT);
  transpose_cvt<<<dim3(DIN / 32, NH / 32, 1),    256, 0, stream>>>(w1, w1t, DIN, NH);
  transpose_cvt<<<dim3(DIN / 32, NH / 32, 1),    256, 0, stream>>>(wg, wgt, DIN, NH);

  router_gemm  <<<B_TOK / 128, 256, 0, stream>>>(xb, w1t, wgt, b1, bg, hbuf);
  logits_kernel<<<B_TOK / 256, 256, 0, stream>>>(hbuf, w2, b2, lbuf);
  softmax_batch<<<NE, 256, 0, stream>>>(lbuf);
  topk_kernel  <<<NE, 1024, 0, stream>>>(lbuf, ibp, nwsp);

  expert_gemm<<<dim3(BK / 128, DOUT / 128, NE), 256, 0, stream>>>(
      xb, webt, be, ibp, nwsp, out, ts);
  combine_kernel<<<8192, 256, 0, stream>>>(out, ts);
}
